// GAT_52158082843451
// MI455X (gfx1250) — matrix operand order repaired and checked by reading, not compiled, not run on hardware
//
#include <hip/hip_runtime.h>
#include <hip/hip_bf16.h>

// ---------------- problem constants (match reference) ----------------
#define F_IN   128
#define D_EMB  64
#define HEADS  4
#define CHANS  256          // HEADS * D_EMB
#define EDIM   20
#define SLOPE  0.2f

// ---------------- WMMA types (gfx1250, wave32) ----------------
typedef __attribute__((ext_vector_type(16))) __bf16 v16bf;
typedef __attribute__((ext_vector_type(8)))  __bf16 v8bf;
typedef __attribute__((ext_vector_type(8)))  float  v8f;

// ---------------- helpers ----------------
__device__ __forceinline__ void atomicMaxF32(float* addr, float val) {
  // sign-aware float max via integer atomics; smax initialized to -inf
  if (val >= 0.0f) atomicMax((int*)addr, __float_as_int(val));
  else             atomicMin((unsigned int*)addr, __float_as_uint(val));
}

// =====================================================================
// Weight prep (once per launch, tiny): Wt[n][k] = bf16(W[k][n]).
// Puts the B-fragment K-run contiguous so the GEMM loads it as v16bf.
// =====================================================================
__global__ void w_transpose_bf16(const float* __restrict__ W, __bf16* __restrict__ Wt,
                                 int K, int Nout) {
  int i = blockIdx.x * blockDim.x + threadIdx.x;
  if (i >= K * Nout) return;
  int k = i / Nout, n = i % Nout;             // coalesced read over n
  Wt[(size_t)n * K + k] = (__bf16)W[i];
}

// =====================================================================
// GEMM 1: h[N,64] = relu(x[N,128] @ f_W + f_b), A fp32, output bf16.
// One wave = 16x64 strip = 4 accumulators sharing one A fragment.
// B fragment: single v16bf load from transposed bf16 weights
//   (lane col = n0+16j+lr, elements t -> K = k0 + t + 16*half, contiguous).
// A fragment: two contiguous 8-float runs per lane, loaded as float4s.
// Contiguous K order (16*half + t) for A and B alike, not the ISA table order.
// =====================================================================
__global__ __launch_bounds__(32)
void gemm_x_to_h(const float* __restrict__ A, const __bf16* __restrict__ Wt,
                 const float* __restrict__ bias, __bf16* __restrict__ C,
                 int K, int Nout) {
  const int lane = threadIdx.x & 31;
  const int half = lane >> 4;
  const int lr   = lane & 15;
  const int m0 = blockIdx.x * 16;
  const int n0 = blockIdx.y * 64;

  v8f acc[4] = {v8f{}, v8f{}, v8f{}, v8f{}};
  for (int k0 = 0; k0 < K; k0 += 32) {
    const float* Ap = A + (size_t)(m0 + lr) * K + k0 + 16 * half;
    float4 a0 = *(const float4*)(Ap + 0);
    float4 a1 = *(const float4*)(Ap + 4);
    float4 a2 = *(const float4*)(Ap + 8);
    float4 a3 = *(const float4*)(Ap + 12);
    v16bf a;
    a[0]  = (__bf16)a0.x; a[1]  = (__bf16)a0.y; a[2]  = (__bf16)a0.z; a[3]  = (__bf16)a0.w;
    a[4]  = (__bf16)a1.x; a[5]  = (__bf16)a1.y; a[6]  = (__bf16)a1.z; a[7]  = (__bf16)a1.w;
    a[8]  = (__bf16)a2.x; a[9]  = (__bf16)a2.y; a[10] = (__bf16)a2.z; a[11] = (__bf16)a2.w;
    a[12] = (__bf16)a3.x; a[13] = (__bf16)a3.y; a[14] = (__bf16)a3.z; a[15] = (__bf16)a3.w;
#pragma unroll
    for (int j = 0; j < 4; ++j) {
      const int col = n0 + 16 * j + lr;
      v16bf b = *(const v16bf*)(Wt + (size_t)col * K + k0 + 16 * half);
      acc[j] = __builtin_amdgcn_wmma_f32_16x16x32_bf16(
          false, a, false, b, (short)0, acc[j], false, false);
    }
  }
#pragma unroll
  for (int j = 0; j < 4; ++j) {
    const int col = n0 + 16 * j + lr;
    const float bv = bias[col];
#pragma unroll
    for (int r = 0; r < 8; ++r) {
      float v = acc[j][r] + bv;
      v = v > 0.0f ? v : 0.0f;                    // relu
      C[(size_t)(m0 + r + 8 * half) * Nout + col] = (__bf16)v;
    }
  }
}

// =====================================================================
// GEMM 2: X[N,256] = h[N,64] @ W + b, A bf16 (h), output fp32 (xl/xr).
// A fragment: two contiguous v8bf (16B) loads, no conversions.
// =====================================================================
__global__ __launch_bounds__(32)
void gemm_h_to_x(const __bf16* __restrict__ A, const __bf16* __restrict__ Wt,
                 const float* __restrict__ bias, float* __restrict__ C,
                 int K, int Nout) {
  const int lane = threadIdx.x & 31;
  const int half = lane >> 4;
  const int lr   = lane & 15;
  const int m0 = blockIdx.x * 16;
  const int n0 = blockIdx.y * 64;

  v8f acc[4] = {v8f{}, v8f{}, v8f{}, v8f{}};
  for (int k0 = 0; k0 < K; k0 += 32) {
    const __bf16* Ap = A + (size_t)(m0 + lr) * K + k0 + 16 * half;
    v8bf lo = *(const v8bf*)(Ap);         // elements 0..7  : K = 16h + t
    v8bf hi = *(const v8bf*)(Ap + 8);     // elements 8..15 : K = 16h + 8 + t
    v16bf a = __builtin_shufflevector(lo, hi, 0, 1, 2, 3, 4, 5, 6, 7,
                                              8, 9, 10, 11, 12, 13, 14, 15);
#pragma unroll
    for (int j = 0; j < 4; ++j) {
      const int col = n0 + 16 * j + lr;
      v16bf b = *(const v16bf*)(Wt + (size_t)col * K + k0 + 16 * half);
      acc[j] = __builtin_amdgcn_wmma_f32_16x16x32_bf16(
          false, a, false, b, (short)0, acc[j], false, false);
    }
  }
#pragma unroll
  for (int j = 0; j < 4; ++j) {
    const int col = n0 + 16 * j + lr;
    const float bv = bias[col];
#pragma unroll
    for (int r = 0; r < 8; ++r)
      C[(size_t)(m0 + r + 8 * half) * Nout + col] = acc[j][r] + bv;
  }
}

// =====================================================================
// e_emb[E,20] = relu(edge_attr[E,2] @ fe_W[2,20] + fe_b)   (once)
// =====================================================================
__global__ void edge_embed(const float* __restrict__ ea, const float* __restrict__ feW,
                           const float* __restrict__ feb, float* __restrict__ eemb, int E) {
  int i = blockIdx.x * blockDim.x + threadIdx.x;
  if (i >= E * EDIM) return;
  int e = i / EDIM, j = i % EDIM;
  float v = fmaf(ea[e * 2 + 0], feW[j], fmaf(ea[e * 2 + 1], feW[EDIM + j], feb[j]));
  eemb[i] = v > 0.0f ? v : 0.0f;
}

__global__ void fill_f32(float* __restrict__ p, float v, int n) {
  int i = blockIdx.x * blockDim.x + threadIdx.x;
  if (i < n) p[i] = v;
}

// =====================================================================
// Pass 1: one wave per edge. score[e,h] = sum_d att[h,d] *
//   leaky( xl[src,h,d] + xr[dst,h,d] + (e_emb[e,:] @ We[:,h*64+d]) )
// ee recomputed on the fly (20 MACs/ch) -> no 819MB [E,256] intermediate.
// xl/xr/aggr are 51MB each -> L2-resident gathers/scatters.
// =====================================================================
__global__ __launch_bounds__(256)
void edge_score(const float* __restrict__ xl, const float* __restrict__ xr,
                const float* __restrict__ eemb, const float* __restrict__ We,
                const float* __restrict__ att, const int* __restrict__ src,
                const int* __restrict__ dst, float* __restrict__ score,
                float* __restrict__ smax, int E) {
  int e = blockIdx.x * (blockDim.x >> 5) + (threadIdx.x >> 5);
  if (e >= E) return;
  const int lane = threadIdx.x & 31;
  const int s = src[e], d = dst[e];

  float e20[EDIM];
#pragma unroll
  for (int k = 0; k < EDIM; ++k) e20[k] = eemb[(size_t)e * EDIM + k];  // broadcast loads

  float hacc[HEADS] = {0.f, 0.f, 0.f, 0.f};
#pragma unroll
  for (int i = 0; i < 8; ++i) {                 // lane owns channels lane+32i
    int ch = lane + 32 * i;                     // head = i>>1
    float ee = 0.0f;
#pragma unroll
    for (int k = 0; k < EDIM; ++k) ee = fmaf(e20[k], We[k * CHANS + ch], ee);
    float m = xl[(size_t)s * CHANS + ch] + xr[(size_t)d * CHANS + ch] + ee;
    m = m > 0.0f ? m : SLOPE * m;
    hacc[i >> 1] = fmaf(m, att[ch], hacc[i >> 1]);
  }
#pragma unroll
  for (int h = 0; h < HEADS; ++h) {
    float v = hacc[h];
#pragma unroll
    for (int off = 16; off; off >>= 1) v += __shfl_xor(v, off, 32);
    if (lane == h) {
      score[(size_t)e * HEADS + h] = v;
      atomicMaxF32(&smax[(size_t)d * HEADS + h], v);
    }
  }
}

// =====================================================================
// Pass 2: ex = exp(score - smax[dst]); score <- ex; denom[dst] += ex
// =====================================================================
__global__ void edge_exp(const int* __restrict__ dst, float* __restrict__ score,
                         const float* __restrict__ smax, float* __restrict__ denom, int E) {
  int i = blockIdx.x * blockDim.x + threadIdx.x;
  if (i >= E * HEADS) return;
  int e = i >> 2, h = i & 3;
  int d = dst[e];
  float ex = __expf(score[i] - smax[d * HEADS + h]);
  score[i] = ex;
  unsafeAtomicAdd(&denom[d * HEADS + h], ex);   // native global_atomic_add_f32
}

// =====================================================================
// Pass 3: one wave per edge. aggr[dst,h,d] += (ex/denom[dst,h]) * xl[src,h,d]
// =====================================================================
__global__ __launch_bounds__(256)
void edge_aggregate(const float* __restrict__ xl, const int* __restrict__ src,
                    const int* __restrict__ dst, const float* __restrict__ score,
                    const float* __restrict__ denom, float* __restrict__ aggr, int E) {
  int e = blockIdx.x * (blockDim.x >> 5) + (threadIdx.x >> 5);
  if (e >= E) return;
  const int lane = threadIdx.x & 31;
  const int s = src[e], d = dst[e];
  float alpha[HEADS];
#pragma unroll
  for (int h = 0; h < HEADS; ++h)
    alpha[h] = score[(size_t)e * HEADS + h] / denom[(size_t)d * HEADS + h];
#pragma unroll
  for (int i = 0; i < 8; ++i) {
    int ch = lane + 32 * i;
    unsafeAtomicAdd(&aggr[(size_t)d * CHANS + ch],
                    alpha[i >> 1] * xl[(size_t)s * CHANS + ch]);
  }
}

// =====================================================================
// h_out = relu(mean_h aggr + bias): bf16 variant (feeds next GEMM)
// and fp32 variant (final layer -> d_out).
// =====================================================================
__global__ void node_finalize_bf16(const float* __restrict__ aggr, const float* __restrict__ bias,
                                   __bf16* __restrict__ hout, int N) {
  int i = blockIdx.x * blockDim.x + threadIdx.x;
  if (i >= N * D_EMB) return;
  int n = i >> 6, dd = i & 63;
  const float* a = aggr + (size_t)n * CHANS + dd;
  float v = 0.25f * (a[0] + a[64] + a[128] + a[192]) + bias[dd];
  hout[i] = (__bf16)(v > 0.0f ? v : 0.0f);
}

__global__ void node_finalize_f32(const float* __restrict__ aggr, const float* __restrict__ bias,
                                  float* __restrict__ hout, int N) {
  int i = blockIdx.x * blockDim.x + threadIdx.x;
  if (i >= N * D_EMB) return;
  int n = i >> 6, dd = i & 63;
  const float* a = aggr + (size_t)n * CHANS + dd;
  float v = 0.25f * (a[0] + a[64] + a[128] + a[192]) + bias[dd];
  hout[i] = v > 0.0f ? v : 0.0f;
}

// =====================================================================
extern "C" void kernel_launch(void* const* d_in, const int* in_sizes, int n_in,
                              void* d_out, int out_size, void* d_ws, size_t ws_size,
                              hipStream_t stream) {
  const float* x    = (const float*)d_in[0];
  const float* ea   = (const float*)d_in[1];
  const int*   ei   = (const int*)  d_in[2];
  const float* fW   = (const float*)d_in[3];
  const float* fb   = (const float*)d_in[4];
  const float* feW  = (const float*)d_in[5];
  const float* feb  = (const float*)d_in[6];
  const float* Wl   = (const float*)d_in[7];
  const float* bl   = (const float*)d_in[8];
  const float* Wr   = (const float*)d_in[9];
  const float* br   = (const float*)d_in[10];
  const float* We   = (const float*)d_in[11];
  const float* att  = (const float*)d_in[12];
  const float* bias = (const float*)d_in[13];
  float* out = (float*)d_out;

  const int N = in_sizes[0] / F_IN;    // 50000
  const int E = in_sizes[1] / 2;       // 800000
  const int* src = ei;
  const int* dst = ei + E;

  // ---- workspace layout (~245 MB), 256B-aligned slots ----
  char* base = (char*)d_ws;
  auto alloc = [&](size_t bytes) -> char* {
    char* q = base;
    base += (bytes + 255) & ~(size_t)255;
    return q;
  };
  float*  xl    = (float*)alloc((size_t)N * CHANS * 4);
  float*  xr    = (float*)alloc((size_t)N * CHANS * 4);
  float*  eemb  = (float*)alloc((size_t)E * EDIM * 4);
  float*  score = (float*)alloc((size_t)E * HEADS * 4);
  float*  smax  = (float*)alloc((size_t)N * HEADS * 4);
  float*  denom = (float*)alloc((size_t)N * HEADS * 4);
  float*  aggr  = (float*)alloc((size_t)N * CHANS * 4);
  __bf16* h_a   = (__bf16*)alloc((size_t)N * D_EMB * 2);
  __bf16* h_b   = (__bf16*)alloc((size_t)N * D_EMB * 2);
  __bf16* Wt_f  = (__bf16*)alloc((size_t)F_IN * D_EMB * 2);   // [64][128]
  __bf16* Wt_l  = (__bf16*)alloc((size_t)D_EMB * CHANS * 2);  // [256][64]
  __bf16* Wt_r  = (__bf16*)alloc((size_t)D_EMB * CHANS * 2);  // [256][64]
  (void)ws_size; (void)n_in; (void)out_size;

  // ---- weight prep (tiny, once per launch) ----
  w_transpose_bf16<<<(F_IN * D_EMB + 255) / 256, 256, 0, stream>>>(fW, Wt_f, F_IN, D_EMB);
  w_transpose_bf16<<<(D_EMB * CHANS + 255) / 256, 256, 0, stream>>>(Wl, Wt_l, D_EMB, CHANS);
  w_transpose_bf16<<<(D_EMB * CHANS + 255) / 256, 256, 0, stream>>>(Wr, Wt_r, D_EMB, CHANS);

  // h = relu(x @ f_W + f_b)  [N,128]x[128,64] -> bf16
  gemm_x_to_h<<<dim3(N / 16, D_EMB / 64), 32, 0, stream>>>(x, Wt_f, fb, h_a, F_IN, D_EMB);

  // e_emb = relu(edge_attr @ fe_W + fe_b)   (constant across layers)
  edge_embed<<<(E * EDIM + 255) / 256, 256, 0, stream>>>(ea, feW, feb, eemb, E);

  const float NEG_INF = -__builtin_inff();
  for (int layer = 0; layer < 3; ++layer) {
    const __bf16* hin = (layer == 1) ? h_b : h_a;

    // xl = h@Wl+bl, xr = h@Wr+br     [N,64]x[64,256] -> fp32
    gemm_h_to_x<<<dim3(N / 16, CHANS / 64), 32, 0, stream>>>(hin, Wt_l, bl, xl, D_EMB, CHANS);
    gemm_h_to_x<<<dim3(N / 16, CHANS / 64), 32, 0, stream>>>(hin, Wt_r, br, xr, D_EMB, CHANS);

    fill_f32<<<(N * HEADS + 255) / 256, 256, 0, stream>>>(smax, NEG_INF, N * HEADS);
    hipMemsetAsync(denom, 0, (size_t)N * HEADS * sizeof(float), stream);
    hipMemsetAsync(aggr,  0, (size_t)N * CHANS * sizeof(float), stream);

    edge_score<<<(E + 7) / 8, 256, 0, stream>>>(xl, xr, eemb, We, att,
                                                src, dst, score, smax, E);
    edge_exp<<<(E * HEADS + 255) / 256, 256, 0, stream>>>(dst, score, smax, denom, E);
    edge_aggregate<<<(E + 7) / 8, 256, 0, stream>>>(xl, src, dst, score, denom, aggr, E);

    if (layer == 2)
      node_finalize_f32<<<(N * D_EMB + 255) / 256, 256, 0, stream>>>(aggr, bias, out, N);
    else
      node_finalize_bf16<<<(N * D_EMB + 255) / 256, 256, 0, stream>>>(
          aggr, bias, (layer == 0) ? h_b : h_a, N);
  }
}
